// GroupConvLayer_13280038879807
// MI455X (gfx1250) — matrix operand order repaired and checked by reading, not compiled, not run on hardware
//
#include <hip/hip_runtime.h>

// GroupConv via orbit-indexed weights, mapped onto CDNA5 WMMA.
// out[b,i,o] = sum_{j,c} x[b,j,c] * W[o,c,pair_orbit[i,j]]  (+ bias[o])
//
//  - V_WMMA_F32_16X16X32_F16, f32 accumulation.
//  - All 24 orbit weight matrices pre-swizzled in LDS as f16 B-fragments (192KB).
//  - Workgroup = 768 threads = 24 waves; wave w = output row i, computing a
//    32-batch x 64-output tile (two M-tiles -> each B-fragment feeds 2 WMMAs).
//  - x tile (32x24x64) staged once per tile in LDS as f16 A-fragments (96KB).
//  - Orbit row preloaded into registers once; j-loop fully unrolled.
//  - Non-temporal policy on streaming x loads and out stores (preserve L2).

typedef __attribute__((ext_vector_type(16))) _Float16 v16h;
typedef __attribute__((ext_vector_type(8)))  float    v8f;
typedef __attribute__((ext_vector_type(4)))  float    v4f;   // native vec for NT loads

#define BATCH   32768
#define PP      24
#define C_IN    64
#define C_OUT   64
#define N_ORB   24
#define BTILE   32
#define NBTILES (BATCH / BTILE)       // 1024

#define THREADS 768                   // 24 waves of 32
#define NWAVES  24

// Fragment = 32 lanes x 16 f16 = 512 halfs.
//  weights: frag id ((k*4 + ot)*2 + kk)          -> 192 frags = 98304 halfs (192KB)
//  x tile : frag id ((j*2 + kk)*2 + mt)          ->  96 frags = 49152 halfs ( 96KB)
#define W_FRAGS   (N_ORB * 4 * 2)     // 192
#define X_FRAGS   (PP * 2 * 2)        // 96
#define ORB_STRIDE (4 * 2 * 512)      // halfs per orbit matrix = 4096
#define LDS_H     ((W_FRAGS + X_FRAGS) * 512)   // 147456 halfs = 288KB

__device__ __forceinline__ v8f wmma_f16(v16h a, v16h b, v8f c) {
    // (neg_a, A, neg_b, B, c_mod, C, reuse_a, reuse_b)
    return __builtin_amdgcn_wmma_f32_16x16x32_f16(false, a, false, b,
                                                  (short)0, c, false, false);
}

__global__ __launch_bounds__(THREADS)
void groupconv_wmma_kernel(const float* __restrict__ x,
                           const float* __restrict__ w,
                           const float* __restrict__ bias,
                           const long long* __restrict__ pair_orbit,
                           float* __restrict__ out)
{
    __shared__ _Float16 lds[LDS_H];
    _Float16* wlds = lds;                      // B-fragments for all orbits
    _Float16* xlds = lds + W_FRAGS * 512;      // A-fragments for current tile

    const int tid  = threadIdx.x;
    const int lane = tid & 31;
    const int wave = tid >> 5;                 // output row i

    // ---------------- one-time: stage all orbit weights as B-fragments -----
    // B layout (16-bit, 32x16 KxN): lane l -> n = l&15, h = l>>4;
// element t -> k_local = (t>=8 ? 16 : 0) + h*8 + (t&7). kk selects c block kk*32.
    for (int chunk = tid; chunk < W_FRAGS * 32; chunk += THREADS) {
        const int cl   = chunk & 31;
        const int frag = chunk >> 5;           // ((k*4+ot)*2+kk)
        const int kk   = frag & 1;
        const int ot   = (frag >> 1) & 3;
        const int k    = frag >> 3;            // orbit id
        const int n    = cl & 15;
        const int h    = cl >> 4;
        const int o    = ot * 16 + n;
        const int c0   = kk * 32 + h * 8;
        // weight[o][c][k] at ((o*C_IN + c)*N_ORB + k), c stride = N_ORB
        const float* src = w + ((size_t)o * C_IN + c0) * N_ORB + k;
        _Float16* dst = wlds + (size_t)chunk * 16;
        #pragma unroll
        for (int t = 0; t < 16; ++t)
            dst[t] = (_Float16)src[(size_t)(t < 8 ? t : t + 8) * N_ORB];
    }

    // ---------------- per-wave: preload orbit row (tile-invariant) ---------
    const int i = wave;
    int kb[PP];                                // orbit base offsets (halfs)
    #pragma unroll
    for (int j = 0; j < PP; ++j)
        kb[j] = ((int)pair_orbit[(size_t)i * PP + j]) * ORB_STRIDE;

    __syncthreads();

    const int n_out = lane & 15;               // output column within o-tile
    const int hh    = lane >> 4;

    // ---------------- grid-stride over 32-batch tiles ----------------------
    for (int bt = blockIdx.x; bt < NBTILES; bt += gridDim.x) {
        const int b0 = bt * BTILE;

        // Stage x[b0:b0+32, :, :] as f16 A-fragments (non-temporal reads).
        // A layout (16-bit, 16x32 MxK): lane l -> m = l&15, h = l>>4;
        // element t -> k = (t>=8 ? 16 : 0) + h*8 + (t&7).
        for (int chunk = tid; chunk < X_FRAGS * 32; chunk += THREADS) {
            const int cl   = chunk & 31;
            const int frag = chunk >> 5;       // ((j*2+kk)*2+mt)
            const int mt   = frag & 1;
            const int kk   = (frag >> 1) & 1;
            const int j    = frag >> 2;
            const int m    = cl & 15;
            const int h    = cl >> 4;
            const int b    = b0 + mt * 16 + m;
            const float* src = x + ((size_t)b * PP + j) * C_IN + kk * 32;
            const v4f* s0 = (const v4f*)(src + h * 8);       // k 0..7
            const v4f* s1 = (const v4f*)(src + 16 + h * 8);  // k 16..23
            v4f p0 = __builtin_nontemporal_load(s0);
            v4f p1 = __builtin_nontemporal_load(s0 + 1);
            v4f p2 = __builtin_nontemporal_load(s1);
            v4f p3 = __builtin_nontemporal_load(s1 + 1);
            _Float16* dst = xlds + (size_t)chunk * 16;
            #pragma unroll
            for (int t = 0; t < 4; ++t) dst[t]      = (_Float16)p0[t];
            #pragma unroll
            for (int t = 0; t < 4; ++t) dst[4 + t]  = (_Float16)p1[t];
            #pragma unroll
            for (int t = 0; t < 4; ++t) dst[8 + t]  = (_Float16)p2[t];
            #pragma unroll
            for (int t = 0; t < 4; ++t) dst[12 + t] = (_Float16)p3[t];
        }
        __syncthreads();

        // ------------- compute: wave `i`, 32 batches x 64 outputs ----------
        v8f acc[2][4];
        #pragma unroll
        for (int mt = 0; mt < 2; ++mt)
            #pragma unroll
            for (int ot = 0; ot < 4; ++ot) acc[mt][ot] = (v8f){};

        #pragma unroll
        for (int j = 0; j < PP; ++j) {
            const _Float16* xb = xlds + (size_t)j * 2048 + (size_t)lane * 16;
            const v16h a00 = *(const v16h*)(xb);          // mt0 kk0
            const v16h a10 = *(const v16h*)(xb + 512);    // mt1 kk0
            const v16h a01 = *(const v16h*)(xb + 1024);   // mt0 kk1
            const v16h a11 = *(const v16h*)(xb + 1536);   // mt1 kk1

            const _Float16* wb = wlds + kb[j] + (size_t)lane * 16;
            #pragma unroll
            for (int ot = 0; ot < 4; ++ot) {
                const v16h bq0 = *(const v16h*)(wb + ot * 1024);        // kk0
                const v16h bq1 = *(const v16h*)(wb + ot * 1024 + 512);  // kk1
                acc[0][ot] = wmma_f16(a00, bq0, acc[0][ot]);
                acc[1][ot] = wmma_f16(a10, bq0, acc[1][ot]);
                acc[0][ot] = wmma_f16(a01, bq1, acc[0][ot]);
                acc[1][ot] = wmma_f16(a11, bq1, acc[1][ot]);
            }
        }

        // ------------- store D (32-bit C/D layout), non-temporal -----------
        // lane l (0-15): N = l, M = r ; lane l+16: N = l, M = 8 + r.
        #pragma unroll
        for (int mt = 0; mt < 2; ++mt) {
            #pragma unroll
            for (int ot = 0; ot < 4; ++ot) {
                const float bv = bias[ot * 16 + n_out];
                #pragma unroll
                for (int r = 0; r < 8; ++r) {
                    const int b = b0 + mt * 16 + hh * 8 + r;
                    float* dp = out + ((size_t)b * PP + i) * C_OUT + ot * 16 + n_out;
                    __builtin_nontemporal_store(acc[mt][ot][r] + bv, dp);
                }
            }
        }
        __syncthreads();   // protect xlds before next tile's staging
    }
}

extern "C" void kernel_launch(void* const* d_in, const int* in_sizes, int n_in,
                              void* d_out, int out_size, void* d_ws, size_t ws_size,
                              hipStream_t stream) {
    const float*     x    = (const float*)d_in[0];
    const float*     w    = (const float*)d_in[1];
    const float*     bias = (const float*)d_in[2];
    const long long* po   = (const long long*)d_in[3];   // int64 orbit table
    float*           out  = (float*)d_out;

    // 512 workgroups x 2 batch tiles each; 288KB LDS -> 1 workgroup per WGP.
    groupconv_wmma_kernel<<<512, THREADS, 0, stream>>>(x, w, bias, po, out);
}
